// VentMamba_3238405341861
// MI455X (gfx1250) — matrix operand order repaired and checked by reading, not compiled, not run on hardware
//
#include <hip/hip_runtime.h>
#include <hip/hip_bf16.h>

typedef __attribute__((ext_vector_type(8)))  _Float16 v8h;
typedef __attribute__((ext_vector_type(16))) _Float16 v16h;
typedef __attribute__((ext_vector_type(8)))  float    v8f;

// ---------------------------------------------------------------------------
// elementwise / layout helpers
// ---------------------------------------------------------------------------
__global__ void k_cvt_f16(const float* __restrict__ src, _Float16* __restrict__ dst, size_t n) {
  size_t i = (size_t)blockIdx.x * blockDim.x + threadIdx.x;
  if (i < n) dst[i] = (_Float16)src[i];
}

// src: K x N (f32 row-major). dst: Np x Kp (f16), dst[n*Kp+k] = src[k*N+n], zero padded.
__global__ void k_transpose_pad(const float* __restrict__ src, _Float16* __restrict__ dst,
                                int K, int N, int Kp, int Np) {
  int i = blockIdx.x * blockDim.x + threadIdx.x;
  int total = Kp * Np;
  if (i >= total) return;
  int n = i / Kp, k = i - n * Kp;
  float v = (n < N && k < K) ? src[(size_t)k * N + n] : 0.0f;
  dst[i] = (_Float16)v;
}

// dbl: M x 48 (cols 0..7 = dt). dt_h: M x 32 f16 (K padded with zeros)
__global__ void k_dt_pack(const float* __restrict__ dbl, _Float16* __restrict__ dth, int M) {
  int i = blockIdx.x * blockDim.x + threadIdx.x;
  if (i >= M * 32) return;
  int k = i & 31; int m = i >> 5;
  float v = (k < 8) ? dbl[(size_t)m * 48 + k] : 0.0f;
  dth[i] = (_Float16)v;
}

// ---------------------------------------------------------------------------
// WMMA GEMM: C[M,N] = A[M,K] * Bt[N,K]^T (+bias). One wave -> 16 x (16*NT) strip.
// A lane layout (16-bit A 16x32): lanes 0-15 K=0..15 contiguous, lanes 16-31 K=16..31
// B lane layout (16-bit B 32x16): lanes 0-15 K=0..15 contiguous, lanes 16-31 K=16..31
// ---------------------------------------------------------------------------
template<int K, int NT>
__global__ void k_gemm(const _Float16* __restrict__ A, const _Float16* __restrict__ Bt,
                       float* __restrict__ C, const float* __restrict__ bias,
                       int M, int N) {
  const int lane = threadIdx.x & 31;
  const int wave = (int)((blockIdx.x * (size_t)blockDim.x + threadIdx.x) >> 5);
  const int ntiles = N / (16 * NT);
  const int mt = wave / ntiles;
  const int nt = wave - mt * ntiles;
  if (mt * 16 >= M) return;
  const int m0 = mt * 16;
  const int n0 = nt * 16 * NT;
  const int r  = lane & 15;
  const int hi = lane >> 4;

  v8f acc[NT] = {};

  const _Float16* ap = A + (size_t)(m0 + r) * K + (hi << 4);
#pragma unroll
  for (int k0 = 0; k0 < K; k0 += 32) {
    v8h alo = *(const v8h*)(ap + k0);
    v8h ahi = *(const v8h*)(ap + k0 + 8);
    v16h av = __builtin_shufflevector(alo, ahi, 0,1,2,3,4,5,6,7,8,9,10,11,12,13,14,15);
#pragma unroll
    for (int t = 0; t < NT; ++t) {
      const _Float16* bp = Bt + (size_t)(n0 + t * 16 + r) * K + k0 + (hi << 4);
      v16h bv = *(const v16h*)bp;
      acc[t] = __builtin_amdgcn_wmma_f32_16x16x32_f16(false, av, false, bv,
                                                      (short)0, acc[t], false, false);
    }
  }
  const int mrow = m0 + (hi << 3);
#pragma unroll
  for (int t = 0; t < NT; ++t) {
    const int cn = n0 + t * 16 + r;
    const float bv = bias ? bias[cn] : 0.0f;
#pragma unroll
    for (int j = 0; j < 8; ++j)
      C[(size_t)(mrow + j) * N + cn] = acc[t][j] + bv;
  }
}

// ---------------------------------------------------------------------------
// LayerNorm (dim=128), wave per row -> f16 output
// ---------------------------------------------------------------------------
__global__ void k_ln0(const float* __restrict__ X, const float* __restrict__ g,
                      const float* __restrict__ b, _Float16* __restrict__ Y, int M) {
  int wave = (int)((blockIdx.x * (size_t)blockDim.x + threadIdx.x) >> 5);
  int lane = threadIdx.x & 31;
  if (wave >= M) return;
  const float* row = X + (size_t)wave * 128;
  float v[4], s = 0.f, s2 = 0.f;
#pragma unroll
  for (int i = 0; i < 4; ++i) { v[i] = row[lane + 32 * i]; s += v[i]; s2 += v[i] * v[i]; }
#pragma unroll
  for (int off = 16; off >= 1; off >>= 1) { s += __shfl_xor(s, off, 32); s2 += __shfl_xor(s2, off, 32); }
  float mu  = s * (1.0f / 128.0f);
  float var = s2 * (1.0f / 128.0f) - mu * mu;
  float inv = rsqrtf(var + 1e-5f);
  _Float16* out = Y + (size_t)wave * 128;
#pragma unroll
  for (int i = 0; i < 4; ++i) {
    int c = lane + 32 * i;
    out[c] = (_Float16)((v[i] - mu) * inv * g[c] + b[c]);
  }
}

// LayerNorm + fused attn/fc dot products; never materializes normalized x
__global__ void k_ln1(const float* __restrict__ X, const float* __restrict__ g,
                      const float* __restrict__ b, const float* __restrict__ attnW,
                      const float* __restrict__ attnB, const float* __restrict__ fcW,
                      float* __restrict__ S, float* __restrict__ R, int M) {
  int wave = (int)((blockIdx.x * (size_t)blockDim.x + threadIdx.x) >> 5);
  int lane = threadIdx.x & 31;
  if (wave >= M) return;
  const float* row = X + (size_t)wave * 128;
  float v[4], s = 0.f, s2 = 0.f;
#pragma unroll
  for (int i = 0; i < 4; ++i) { v[i] = row[lane + 32 * i]; s += v[i]; s2 += v[i] * v[i]; }
#pragma unroll
  for (int off = 16; off >= 1; off >>= 1) { s += __shfl_xor(s, off, 32); s2 += __shfl_xor(s2, off, 32); }
  float mu  = s * (1.0f / 128.0f);
  float var = s2 * (1.0f / 128.0f) - mu * mu;
  float inv = rsqrtf(var + 1e-5f);
  float sd = 0.f, rd = 0.f;
#pragma unroll
  for (int i = 0; i < 4; ++i) {
    int c = lane + 32 * i;
    float y = (v[i] - mu) * inv * g[c] + b[c];
    sd += y * attnW[c];
    rd += y * fcW[c];
  }
#pragma unroll
  for (int off = 16; off >= 1; off >>= 1) { sd += __shfl_xor(sd, off, 32); rd += __shfl_xor(rd, off, 32); }
  if (lane == 0) { S[wave] = sd + attnB[0]; R[wave] = rd; }
}

// ---------------------------------------------------------------------------
// depthwise causal conv (k=4) over xc = xz[:, :256], then SiLU -> u (f32 + f16)
// ---------------------------------------------------------------------------
__global__ void k_conv_silu(const float* __restrict__ xz, const float* __restrict__ cw,
                            const float* __restrict__ cb, float* __restrict__ u,
                            _Float16* __restrict__ uh, int L, size_t total) {
  size_t i = (size_t)blockIdx.x * blockDim.x + threadIdx.x;
  if (i >= total) return;
  int d = (int)(i & 255);
  size_t ml = i >> 8;                      // b*L + l
  size_t bb = ml / (size_t)L;
  int l = (int)(ml - bb * (size_t)L);
  float acc = cb[d];
#pragma unroll
  for (int j = 0; j < 4; ++j) {
    int t = l - 3 + j;
    if (t >= 0) acc += xz[((size_t)(ml - 3 + j)) * 512 + d] * cw[d * 4 + j];
  }
  float s = acc / (1.0f + __expf(-acc));   // SiLU
  u[i] = s;
  uh[i] = (_Float16)s;
}

// ---------------------------------------------------------------------------
// selective scan: one lane per (b, d) channel; h[16] in registers.
// B_t/C_t staged through LDS in 64-timestep chunks. Fuses softplus, +u*D,
// *silu(z) and f16 conversion for the output GEMM.
// ---------------------------------------------------------------------------
__global__ void k_scan(const float* __restrict__ u, const float* __restrict__ draw,
                       const float* __restrict__ dbl, const float* __restrict__ xz,
                       const float* __restrict__ A_log, const float* __restrict__ Dp,
                       _Float16* __restrict__ yzh, int L) {
  __shared__ float lB[64][16];
  __shared__ float lC[64][16];
  const int b = blockIdx.x;
  const int d = blockIdx.y * 32 + threadIdx.x;     // blockDim.x == 32
  float a[16], h[16];
#pragma unroll
  for (int n = 0; n < 16; ++n) { a[n] = -__expf(A_log[d * 16 + n]); h[n] = 0.f; }
  const float Dd = Dp[d];
  const size_t bL = (size_t)b * L;
  for (int c0 = 0; c0 < L; c0 += 64) {
    __syncthreads();
    for (int i = threadIdx.x; i < 64 * 16; i += 32) {
      int t = i >> 4, n = i & 15;
      size_t rb = (bL + c0 + t) * 48;
      lB[t][n] = dbl[rb + 8 + n];
      lC[t][n] = dbl[rb + 24 + n];
    }
    __syncthreads();
    for (int t = 0; t < 64; ++t) {
      size_t idx = bL + c0 + t;
      float x = draw[idx * 256 + d];
      float delta = (x > 20.f) ? x : __logf(1.0f + __expf(x));   // softplus
      float uv = u[idx * 256 + d];
      float du = delta * uv;
      float y = 0.f;
#pragma unroll
      for (int n = 0; n < 16; ++n) {
        float dA = __expf(delta * a[n]);
        h[n] = dA * h[n] + du * lB[t][n];
        y += h[n] * lC[t][n];
      }
      float zv = xz[idx * 512 + 256 + d];
      float sil = zv / (1.0f + __expf(-zv));
      yzh[idx * 256 + d] = (_Float16)((y + uv * Dd) * sil);
    }
  }
}

// ---------------------------------------------------------------------------
// softmax pooling over L + final fc: out[b] = sum_l softmax(s)_l * r_l + fc_b
// ---------------------------------------------------------------------------
__global__ void k_pool(const float* __restrict__ S, const float* __restrict__ R,
                       const float* __restrict__ fcB, float* __restrict__ out, int L) {
  __shared__ float red[256];
  const int b = blockIdx.x, tid = threadIdx.x;
  const float* s = S + (size_t)b * L;
  const float* r = R + (size_t)b * L;
  float m = -3.4e38f;
  for (int i = tid; i < L; i += 256) m = fmaxf(m, s[i]);
  red[tid] = m; __syncthreads();
  for (int off = 128; off >= 1; off >>= 1) {
    if (tid < off) red[tid] = fmaxf(red[tid], red[tid + off]);
    __syncthreads();
  }
  m = red[0]; __syncthreads();
  float se = 0.f, sr = 0.f;
  for (int i = tid; i < L; i += 256) {
    float e = __expf(s[i] - m);
    se += e; sr += e * r[i];
  }
  red[tid] = se; __syncthreads();
  for (int off = 128; off >= 1; off >>= 1) { if (tid < off) red[tid] += red[tid + off]; __syncthreads(); }
  se = red[0]; __syncthreads();
  red[tid] = sr; __syncthreads();
  for (int off = 128; off >= 1; off >>= 1) { if (tid < off) red[tid] += red[tid + off]; __syncthreads(); }
  if (tid == 0) out[b] = red[0] / se + fcB[0];
}

// ---------------------------------------------------------------------------
extern "C" void kernel_launch(void* const* d_in, const int* in_sizes, int n_in,
                              void* d_out, int out_size, void* d_ws, size_t ws_size,
                              hipStream_t stream) {
  (void)in_sizes; (void)n_in; (void)out_size; (void)ws_size;
  const int Bsz = 8, L = 4096, M = Bsz * L;   // 32768 rows

  const float* xv      = (const float*)d_in[0];
  const float* in_W    = (const float*)d_in[1];
  const float* in_b    = (const float*)d_in[2];
  const float* ln0_g   = (const float*)d_in[3];
  const float* ln0_b   = (const float*)d_in[4];
  const float* m_inW   = (const float*)d_in[5];
  const float* conv_w  = (const float*)d_in[6];
  const float* conv_b  = (const float*)d_in[7];
  const float* xproj_W = (const float*)d_in[8];
  const float* dt_W    = (const float*)d_in[9];
  const float* dt_b    = (const float*)d_in[10];
  const float* A_log   = (const float*)d_in[11];
  const float* D       = (const float*)d_in[12];
  const float* out_W   = (const float*)d_in[13];
  const float* ln1_g   = (const float*)d_in[14];
  const float* ln1_b   = (const float*)d_in[15];
  const float* attn_W  = (const float*)d_in[16];
  const float* attn_b  = (const float*)d_in[17];
  const float* fc_W    = (const float*)d_in[18];
  const float* fc_b    = (const float*)d_in[19];
  float* out = (float*)d_out;

  // carve scratch (~223 MB)
  char* w = (char*)d_ws;
  size_t off = 0;
  auto carve = [&](size_t bytes) -> void* {
    void* p = w + off; off = (off + bytes + 255) & ~(size_t)255; return p;
  };
  _Float16* xv_h  = (_Float16*)carve((size_t)M * 64 * 2);
  _Float16* inWt  = (_Float16*)carve((size_t)128 * 64 * 2);
  _Float16* minWt = (_Float16*)carve((size_t)512 * 128 * 2);
  _Float16* xprWt = (_Float16*)carve((size_t)48 * 256 * 2);
  _Float16* dtWt  = (_Float16*)carve((size_t)256 * 32 * 2);
  _Float16* outWt = (_Float16*)carve((size_t)128 * 256 * 2);
  float*    x0    = (float*)   carve((size_t)M * 128 * 4);
  _Float16* x_h   = (_Float16*)carve((size_t)M * 128 * 2);
  float*    xz    = (float*)   carve((size_t)M * 512 * 4);
  float*    u     = (float*)   carve((size_t)M * 256 * 4);
  _Float16* u_h   = (_Float16*)carve((size_t)M * 256 * 2);
  float*    dbl   = (float*)   carve((size_t)M * 48 * 4);
  _Float16* dt_h  = (_Float16*)carve((size_t)M * 32 * 2);
  float*    draw  = (float*)   carve((size_t)M * 256 * 4);
  _Float16* yz_h  = (_Float16*)carve((size_t)M * 256 * 2);
  float*    mo    = (float*)   carve((size_t)M * 128 * 4);
  float*    Sl    = (float*)   carve((size_t)M * 4);
  float*    Rl    = (float*)   carve((size_t)M * 4);

  // f16 conversions / weight transposes
  { size_t n = (size_t)M * 64;
    k_cvt_f16<<<(int)((n + 255) / 256), 256, 0, stream>>>(xv, xv_h, n); }
  k_transpose_pad<<<(128*64   + 255)/256, 256, 0, stream>>>(in_W,    inWt,   64, 128,  64, 128);
  k_transpose_pad<<<(512*128  + 255)/256, 256, 0, stream>>>(m_inW,   minWt, 128, 512, 128, 512);
  k_transpose_pad<<<(48*256   + 255)/256, 256, 0, stream>>>(xproj_W, xprWt, 256,  40, 256,  48);
  k_transpose_pad<<<(256*32   + 255)/256, 256, 0, stream>>>(dt_W,    dtWt,    8, 256,  32, 256);
  k_transpose_pad<<<(128*256  + 255)/256, 256, 0, stream>>>(out_W,   outWt, 256, 128, 256, 128);

  auto ggrid = [&](int Nn, int NT) { int waves = (M / 16) * (Nn / (16 * NT)); return (waves + 7) / 8; };

  // G1: x0 = xv @ in_W + in_b
  k_gemm<64, 4><<<ggrid(128, 4), 256, 0, stream>>>(xv_h, inWt, x0, in_b, M, 128);
  // LN0 -> x_h (f16)
  k_ln0<<<(M + 7) / 8, 256, 0, stream>>>(x0, ln0_g, ln0_b, x_h, M);
  // G2: xz = x @ m_inW  (xc = cols 0..255, z = cols 256..511)
  k_gemm<128, 4><<<ggrid(512, 4), 256, 0, stream>>>(x_h, minWt, xz, nullptr, M, 512);
  // depthwise conv + SiLU -> u
  { size_t n = (size_t)M * 256;
    k_conv_silu<<<(int)((n + 255) / 256), 256, 0, stream>>>(xz, conv_w, conv_b, u, u_h, L, n); }
  // G3: dbl = u @ xproj_W  (dt | B | C, N padded to 48)
  k_gemm<256, 3><<<ggrid(48, 3), 256, 0, stream>>>(u_h, xprWt, dbl, nullptr, M, 48);
  // pack dt to padded f16
  k_dt_pack<<<(M * 32 + 255) / 256, 256, 0, stream>>>(dbl, dt_h, M);
  // G4: draw = dt @ dt_W + dt_b  (softplus applied inside scan)
  k_gemm<32, 4><<<ggrid(256, 4), 256, 0, stream>>>(dt_h, dtWt, draw, dt_b, M, 256);
  // selective scan (fuses softplus, +u*D, *silu(z), f16 cast)
  { dim3 sg(Bsz, 8);
    k_scan<<<sg, 32, 0, stream>>>(u, draw, dbl, xz, A_log, D, yz_h, L); }
  // G5: mo = yz @ out_W
  k_gemm<256, 4><<<ggrid(128, 4), 256, 0, stream>>>(yz_h, outWt, mo, nullptr, M, 128);
  // LN1 + fused attn/fc dots
  k_ln1<<<(M + 7) / 8, 256, 0, stream>>>(mo, ln1_g, ln1_b, attn_W, attn_b, fc_W, Sl, Rl, M);
  // softmax pooling + fc bias
  k_pool<<<Bsz, 256, 0, stream>>>(Sl, Rl, fc_b, out, L);
}
